// Attention_19980187861444
// MI455X (gfx1250) — matrix operand order repaired and checked by reading, not compiled, not run on hardware
//
#include <hip/hip_runtime.h>

// Problem constants (fixed by the reference)
#define BB 4
#define NN 2048
#define CC 1024
#define HH 16
#define DD 64
#define MM (BB*NN)       // 8192 rows of x
#define N3C (3*CC)       // 3072

typedef __bf16 bf16;
typedef __attribute__((ext_vector_type(16))) __bf16 bf16x16;
typedef __attribute__((ext_vector_type(8)))  __bf16 bf16x8;
typedef __attribute__((ext_vector_type(8)))  float  f32x8;

union Frag16 { bf16x16 v; bf16x8 h[2]; };

__device__ __forceinline__ f32x8 wmma_bf16(bf16x16 a, bf16x16 b, f32x8 c) {
    return __builtin_amdgcn_wmma_f32_16x16x32_bf16(
        false, a, false, b, (short)0, c, false, false);
}

__device__ __forceinline__ bf16x8 cvt8(float4 lo, float4 hi) {
    bf16x8 r;
    r[0]=(bf16)lo.x; r[1]=(bf16)lo.y; r[2]=(bf16)lo.z; r[3]=(bf16)lo.w;
    r[4]=(bf16)hi.x; r[5]=(bf16)hi.y; r[6]=(bf16)hi.z; r[7]=(bf16)hi.w;
    return r;
}

// ---------------- weight fp32 -> bf16 conversion ----------------
__global__ __launch_bounds__(256) void cvt_kernel(const float* __restrict__ in,
                                                  unsigned short* __restrict__ out,
                                                  int n) {
    bf16* o = (bf16*)out;
    int i = blockIdx.x * blockDim.x + threadIdx.x;
    int stride = gridDim.x * blockDim.x;
    for (; i < n; i += stride) o[i] = (bf16)in[i];
}

// ---------------- QKV GEMM: qkv = x @ qkv_w^T, scatter to Q/K/V ----------------
// Block tile 128(M) x 128(N); 8 waves; register double-buffered k-pipeline.
__global__ __launch_bounds__(256) void qkv_gemm_kernel(
    const float* __restrict__ x, const unsigned short* __restrict__ w_,
    unsigned short* __restrict__ q_, unsigned short* __restrict__ k_,
    unsigned short* __restrict__ v_) {
    const bf16* w = (const bf16*)w_;
    const int lane = threadIdx.x & 31, wv = threadIdx.x >> 5;
    const int lane15 = lane & 15, khalf = lane >> 4;
    const int mbase = blockIdx.y * 128 + wv * 16;
    const int nbase = blockIdx.x * 128;

    const float* ap = x + (size_t)(mbase + lane15) * CC + khalf * 8;
    const bf16*  bp = w + (size_t)(nbase + lane15) * CC + khalf * 8;

    float4 a0[4], a1[4];
    bf16x8 b0v[8][2], b1v[8][2];

    auto loadA = [&](float4 (&buf)[4], int off) {
        const float* p = ap + off;
        buf[0] = *(const float4*)(p);      buf[1] = *(const float4*)(p + 4);
        buf[2] = *(const float4*)(p + 16); buf[3] = *(const float4*)(p + 20);
    };
    auto loadB = [&](bf16x8 (&buf)[8][2], int off) {
        #pragma unroll
        for (int t = 0; t < 8; ++t) {
            const bf16* p = bp + (size_t)t * (16 * CC) + off;
            buf[t][0] = *(const bf16x8*)(p);
            buf[t][1] = *(const bf16x8*)(p + 16);
        }
    };

    f32x8 acc[8] = {};
    loadA(a0, 0); loadB(b0v, 0);
    for (int kc = 0; kc < CC; kc += 64) {
        // prefetch second half of this 64-wide k-step
        loadA(a1, kc + 32); loadB(b1v, kc + 32);
        Frag16 af; af.h[0] = cvt8(a0[0], a0[1]); af.h[1] = cvt8(a0[2], a0[3]);
        #pragma unroll
        for (int t = 0; t < 8; ++t) {
            Frag16 bf_; bf_.h[0] = b0v[t][0]; bf_.h[1] = b0v[t][1];
            acc[t] = wmma_bf16(af.v, bf_.v, acc[t]);
        }
        // prefetch first half of next 64-wide k-step
        if (kc + 64 < CC) { loadA(a0, kc + 64); loadB(b0v, kc + 64); }
        Frag16 ag; ag.h[0] = cvt8(a1[0], a1[1]); ag.h[1] = cvt8(a1[2], a1[3]);
        #pragma unroll
        for (int t = 0; t < 8; ++t) {
            Frag16 bg; bg.h[0] = b1v[t][0]; bg.h[1] = b1v[t][1];
            acc[t] = wmma_bf16(ag.v, bg.v, acc[t]);
        }
    }

    // Scatter into Q/K/V [B,H,N,D] bf16 (Q pre-scaled by D^-0.5 = 0.125)
    const int which = nbase / CC;
    const float mul = (which == 0) ? 0.125f : 1.0f;
    bf16* dst = (which == 0) ? (bf16*)q_ : (which == 1) ? (bf16*)k_ : (bf16*)v_;
    #pragma unroll
    for (int t = 0; t < 8; ++t) {
        int c = (nbase + t * 16 + lane15) & (CC - 1);
        int h = c >> 6, d = c & 63;
        #pragma unroll
        for (int r = 0; r < 8; ++r) {
            int m = mbase + khalf * 8 + r;
            int b = m >> 11, n = m & (NN - 1);
            dst[(((size_t)(b * HH + h) * NN) + n) * DD + d] = (bf16)(acc[t][r] * mul);
        }
    }
}

// ---------------- Flash attention over [B,H,N,D] bf16 ----------------
// grid.x = N/128 (q blocks), grid.y = B*H; 64-key blocks, double-buffered LDS.
__global__ __launch_bounds__(256) void attn_kernel(
    const unsigned short* __restrict__ q_, const unsigned short* __restrict__ k_,
    const unsigned short* __restrict__ v_, unsigned short* __restrict__ o_) {
    __shared__ __attribute__((aligned(16))) bf16 Ks[2][64 * 64];   // [key][d]
    __shared__ __attribute__((aligned(16))) bf16 Vt[2][64 * 64];   // [d][key]
    __shared__ __attribute__((aligned(16))) bf16 Ps[8 * 16 * 64];  // per-wave P

    const int bh = blockIdx.y;
    const bf16* Qb = (const bf16*)q_ + (size_t)bh * NN * DD;
    const bf16* Kb = (const bf16*)k_ + (size_t)bh * NN * DD;
    const bf16* Vb = (const bf16*)v_ + (size_t)bh * NN * DD;

    const int lane = threadIdx.x & 31, wv = threadIdx.x >> 5;
    const int lane15 = lane & 15, khalf = lane >> 4;

    // Q fragments (A layout), D=64 contraction in two 32-wide chunks
    const int qrow = blockIdx.x * 128 + wv * 16 + lane15;
    const bf16* qp = Qb + (size_t)qrow * DD + khalf * 8;
    Frag16 qf0, qf1;
    qf0.h[0] = *(const bf16x8*)(qp);      qf0.h[1] = *(const bf16x8*)(qp + 16);
    qf1.h[0] = *(const bf16x8*)(qp + 32); qf1.h[1] = *(const bf16x8*)(qp + 48);

    float mrow[8], lrow[8];
    #pragma unroll
    for (int r = 0; r < 8; ++r) { mrow[r] = -1e30f; lrow[r] = 0.0f; }
    f32x8 oacc[4] = {};
    bf16* pw = &Ps[wv * (16 * 64)];

    // staging: thread -> (key row, 16-d chunk)
    const int skey = threadIdx.x >> 2;        // 0..63
    const int sdc  = (threadIdx.x & 3) * 16;  // 0,16,32,48
    bf16x8 kr0, kr1, vr0, vr1;
    auto gload = [&](int kb) {
        const bf16* kp = Kb + (size_t)(kb + skey) * DD + sdc;
        kr0 = *(const bf16x8*)(kp); kr1 = *(const bf16x8*)(kp + 8);
        const bf16* vp = Vb + (size_t)(kb + skey) * DD + sdc;
        vr0 = *(const bf16x8*)(vp); vr1 = *(const bf16x8*)(vp + 8);
    };

    const int J = NN / 64;   // 32 key blocks
    gload(0);
    for (int j = 0; j < J; ++j) {
        const int p = j & 1;
        // stage K and transposed V into buffer p
        *(bf16x8*)&Ks[p][skey * 64 + sdc] = kr0;
        *(bf16x8*)&Ks[p][skey * 64 + sdc + 8] = kr1;
        #pragma unroll
        for (int e = 0; e < 8; ++e) Vt[p][(sdc + e) * 64 + skey] = vr0[e];
        #pragma unroll
        for (int e = 0; e < 8; ++e) Vt[p][(sdc + 8 + e) * 64 + skey] = vr1[e];
        __syncthreads();
        if (j + 1 < J) gload((j + 1) * 64);   // in flight under the math below

        // S = Q * K^T for four 16-key subtiles
        f32x8 S[4];
        #pragma unroll
        for (int s = 0; s < 4; ++s) {
            const bf16* kp = &Ks[p][(s * 16 + lane15) * 64 + khalf * 8];
            Frag16 b0, b1;
            b0.h[0] = *(const bf16x8*)(kp);      b0.h[1] = *(const bf16x8*)(kp + 16);
            b1.h[0] = *(const bf16x8*)(kp + 32); b1.h[1] = *(const bf16x8*)(kp + 48);
            f32x8 c = {};
            c = wmma_bf16(qf0.v, b0.v, c);
            c = wmma_bf16(qf1.v, b1.v, c);
            S[s] = c;
        }

        // Online softmax (rows live in 16-lane halves)
        float alpha[8];
        #pragma unroll
        for (int r = 0; r < 8; ++r) {
            float mx = fmaxf(fmaxf(S[0][r], S[1][r]), fmaxf(S[2][r], S[3][r]));
            #pragma unroll
            for (int m = 1; m < 16; m <<= 1) mx = fmaxf(mx, __shfl_xor(mx, m, 32));
            float mn = fmaxf(mrow[r], mx);
            alpha[r] = __expf(mrow[r] - mn);
            mrow[r] = mn;
            float rs = 0.0f;
            #pragma unroll
            for (int s = 0; s < 4; ++s) {
                float pv = __expf(S[s][r] - mn);
                S[s][r] = pv; rs += pv;
            }
            #pragma unroll
            for (int m = 1; m < 16; m <<= 1) rs += __shfl_xor(rs, m, 32);
            lrow[r] = lrow[r] * alpha[r] + rs;
        }
        #pragma unroll
        for (int t = 0; t < 4; ++t)
            #pragma unroll
            for (int r = 0; r < 8; ++r) oacc[t][r] *= alpha[r];

        // P: C-layout f32 -> row-major bf16 LDS -> A fragments
        #pragma unroll
        for (int s = 0; s < 4; ++s)
            #pragma unroll
            for (int r = 0; r < 8; ++r)
                pw[(khalf * 8 + r) * 64 + s * 16 + lane15] = (bf16)S[s][r];
        __syncthreads();
        const bf16* pp = pw + lane15 * 64 + khalf * 8;
        Frag16 pf0, pf1;
        pf0.h[0] = *(const bf16x8*)(pp);      pf0.h[1] = *(const bf16x8*)(pp + 16);
        pf1.h[0] = *(const bf16x8*)(pp + 32); pf1.h[1] = *(const bf16x8*)(pp + 48);

        // O += P * V  (4 d-tiles x 2 key chunks)
        #pragma unroll
        for (int t = 0; t < 4; ++t) {
            const bf16* vp = &Vt[p][(t * 16 + lane15) * 64 + khalf * 8];
            Frag16 vf0, vf1;
            vf0.h[0] = *(const bf16x8*)(vp);      vf0.h[1] = *(const bf16x8*)(vp + 16);
            vf1.h[0] = *(const bf16x8*)(vp + 32); vf1.h[1] = *(const bf16x8*)(vp + 48);
            oacc[t] = wmma_bf16(pf0.v, vf0.v, oacc[t]);
            oacc[t] = wmma_bf16(pf1.v, vf1.v, oacc[t]);
        }
    }

    // Normalize and store to attention output [B,N,C] bf16
    const int b = bh >> 4, h = bh & 15;
    bf16* O = (bf16*)o_;
    #pragma unroll
    for (int r = 0; r < 8; ++r) {
        float inv = 1.0f / lrow[r];
        int n = blockIdx.x * 128 + wv * 16 + khalf * 8 + r;
        size_t base = ((size_t)(b * NN + n)) * CC + h * DD;
        #pragma unroll
        for (int t = 0; t < 4; ++t)
            O[base + t * 16 + lane15] = (bf16)(oacc[t][r] * inv);
    }
}

// ---------------- Projection GEMM: out = attn @ proj_w^T + b (fp32 out) ----------------
__global__ __launch_bounds__(256) void proj_gemm_kernel(
    const unsigned short* __restrict__ a_, const unsigned short* __restrict__ w_,
    const float* __restrict__ bias, float* __restrict__ out) {
    const bf16* A = (const bf16*)a_;
    const bf16* w = (const bf16*)w_;
    const int lane = threadIdx.x & 31, wv = threadIdx.x >> 5;
    const int lane15 = lane & 15, khalf = lane >> 4;
    const int mbase = blockIdx.y * 128 + wv * 16;
    const int nbase = blockIdx.x * 128;

    const bf16* ap = A + (size_t)(mbase + lane15) * CC + khalf * 8;
    const bf16* bp = w + (size_t)(nbase + lane15) * CC + khalf * 8;

    bf16x8 a0[2], a1[2], b0v[8][2], b1v[8][2];
    auto loadA = [&](bf16x8 (&buf)[2], int off) {
        buf[0] = *(const bf16x8*)(ap + off);
        buf[1] = *(const bf16x8*)(ap + off + 16);
    };
    auto loadB = [&](bf16x8 (&buf)[8][2], int off) {
        #pragma unroll
        for (int t = 0; t < 8; ++t) {
            const bf16* p = bp + (size_t)t * (16 * CC) + off;
            buf[t][0] = *(const bf16x8*)(p);
            buf[t][1] = *(const bf16x8*)(p + 16);
        }
    };

    f32x8 acc[8] = {};
    loadA(a0, 0); loadB(b0v, 0);
    for (int kc = 0; kc < CC; kc += 64) {
        loadA(a1, kc + 32); loadB(b1v, kc + 32);
        Frag16 af; af.h[0] = a0[0]; af.h[1] = a0[1];
        #pragma unroll
        for (int t = 0; t < 8; ++t) {
            Frag16 bf_; bf_.h[0] = b0v[t][0]; bf_.h[1] = b0v[t][1];
            acc[t] = wmma_bf16(af.v, bf_.v, acc[t]);
        }
        if (kc + 64 < CC) { loadA(a0, kc + 64); loadB(b0v, kc + 64); }
        Frag16 ag; ag.h[0] = a1[0]; ag.h[1] = a1[1];
        #pragma unroll
        for (int t = 0; t < 8; ++t) {
            Frag16 bg; bg.h[0] = b1v[t][0]; bg.h[1] = b1v[t][1];
            acc[t] = wmma_bf16(ag.v, bg.v, acc[t]);
        }
    }
    #pragma unroll
    for (int t = 0; t < 8; ++t) {
        int ncol = nbase + t * 16 + lane15;
        float bv = bias[ncol];
        #pragma unroll
        for (int r = 0; r < 8; ++r) {
            int m = mbase + khalf * 8 + r;
            out[(size_t)m * CC + ncol] = acc[t][r] + bv;
        }
    }
}

extern "C" void kernel_launch(void* const* d_in, const int* in_sizes, int n_in,
                              void* d_out, int out_size, void* d_ws, size_t ws_size,
                              hipStream_t stream) {
    (void)in_sizes; (void)n_in; (void)out_size; (void)ws_size;
    const float* x     = (const float*)d_in[0];
    const float* qkvw  = (const float*)d_in[1];
    const float* projw = (const float*)d_in[2];
    const float* projb = (const float*)d_in[3];
    float* out = (float*)d_out;

    // Workspace layout (bf16 elements): qkv_w | proj_w | Q | K | V | attn_out
    unsigned short* ws = (unsigned short*)d_ws;
    unsigned short* wq = ws;
    unsigned short* wp = wq + (size_t)3 * CC * CC;
    unsigned short* Qd = wp + (size_t)CC * CC;
    unsigned short* Kd = Qd + (size_t)MM * CC;
    unsigned short* Vd = Kd + (size_t)MM * CC;
    unsigned short* Ad = Vd + (size_t)MM * CC;

    cvt_kernel<<<2048, 256, 0, stream>>>(qkvw, wq, 3 * CC * CC);
    cvt_kernel<<<1024, 256, 0, stream>>>(projw, wp, CC * CC);
    qkv_gemm_kernel<<<dim3(N3C / 128, MM / 128), 256, 0, stream>>>(x, wq, Qd, Kd, Vd);
    attn_kernel<<<dim3(NN / 128, BB * HH), 256, 0, stream>>>(Qd, Kd, Vd, Ad);
    proj_gemm_kernel<<<dim3(CC / 128, MM / 128), 256, 0, stream>>>(Ad, wp, projb, out);
}
